// GCN_sim_23922967839148
// MI455X (gfx1250) — matrix operand order repaired and checked by reading, not compiled, not run on hardware
//
#include <hip/hip_runtime.h>
#include <hip/hip_bf16.h>

#define BQ 8
#define NN 2048
#define DD 512
#define DQv 128
#define ROWS (BQ * NN)  // 16384

typedef __attribute__((ext_vector_type(16))) _Float16 v16h;
typedef __attribute__((ext_vector_type(8)))  _Float16 v8h;
typedef __attribute__((ext_vector_type(8)))  float    v8f;

__device__ __forceinline__ v8f wmma_f16(v16h a, v16h b, v8f c) {
  return __builtin_amdgcn_wmma_f32_16x16x32_f16(false, a, false, b, (short)0, c,
                                                false, false);
}

// A operand (16x32 f16) built from global row-major data:
// lanes 0-15: M=lane, K=0..15; lanes 16-31: M=lane-16, K=16..31
// -> two contiguous v8h loads per lane. p points at (row m, k0) for this lane.
__device__ __forceinline__ v16h gfragA(const _Float16* p, int hi) {
  union { v16h v; struct { v8h lo; v8h hi8; } s; } u;
  u.s.lo  = *(const v8h*)(p + hi * 16);
  u.s.hi8 = *(const v8h*)(p + 8 + hi * 16);
  return u.v;
}

// A operand gather from LDS (used only for the P transpose in k_attn)
__device__ __forceinline__ v16h ldsA16x32(const _Float16* base, int stride, int lane) {
  const int m = lane & 15;
  const int off = (lane >> 4) << 4;
  v16h a;
#pragma unroll
  for (int h = 0; h < 16; ++h) {
    int k = h + off;
    a[h] = base[m * stride + k];
  }
  return a;
}

// ---------------------------------------------------------------- prep: weights -> f16
__global__ void k_prep(const float* __restrict__ Wf, const float* __restrict__ Wg,
                       _Float16* __restrict__ Wf_h, _Float16* __restrict__ Wg_h) {
  int i = blockIdx.x * blockDim.x + threadIdx.x;
  int stride = gridDim.x * blockDim.x;
  for (int j = i; j < DQv * DD; j += stride) Wf_h[j] = (_Float16)Wf[j];
  for (int j = i; j < DD * DD; j += stride) Wg_h[j] = (_Float16)Wg[j];
}

// ---------------------------------------------------------------- LayerNorm -> xn (f16)
__global__ void __launch_bounds__(128) k_layernorm(const float* __restrict__ x,
                                                   const float* __restrict__ gamma,
                                                   const float* __restrict__ beta,
                                                   _Float16* __restrict__ xn) {
  __shared__ float rs[128], rq[128];
  const int row = blockIdx.x;
  const int tid = threadIdx.x;
  const float4 v = ((const float4*)(x + (size_t)row * DD))[tid];
  rs[tid] = v.x + v.y + v.z + v.w;
  rq[tid] = v.x * v.x + v.y * v.y + v.z * v.z + v.w * v.w;
  __syncthreads();
#pragma unroll
  for (int off = 64; off > 0; off >>= 1) {
    if (tid < off) { rs[tid] += rs[tid + off]; rq[tid] += rq[tid + off]; }
    __syncthreads();
  }
  const float mu  = rs[0] * (1.0f / DD);
  const float var = rq[0] * (1.0f / DD) - mu * mu;
  const float inv = rsqrtf(var + 1e-5f);
  const float xv[4] = {v.x, v.y, v.z, v.w};
#pragma unroll
  for (int j = 0; j < 4; ++j) {
    int c = tid * 4 + j;
    xn[(size_t)row * DD + c] = (_Float16)(((xv[j] - mu) * inv) * gamma[c] + beta[c]);
  }
}

// ---------------------------------------------------------------- Xp = xn*Wf^T + bf, mask, L2-normalize -> Xu (f16)
// All GEMM operands straight from global; LDS only for the cross-lane L2 norm.
__global__ void __launch_bounds__(128) k_proj_norm(const _Float16* __restrict__ xn,
                                                   const _Float16* __restrict__ Wf_h,
                                                   const float* __restrict__ bf,
                                                   const int* __restrict__ mask,
                                                   _Float16* __restrict__ Xu) {
  __shared__ float outT[16 * 132];
  __shared__ float part[16][8];
  __shared__ float invn[16];
  const int tid  = threadIdx.x;
  const int lane = tid & 31;
  const int w    = tid >> 5;
  const int row0 = blockIdx.x * 16;
  const int lane15 = lane & 15;
  const int hi     = lane >> 4;

  const _Float16* arow = xn + (size_t)(row0 + lane15) * DD;
  const _Float16* b0p  = Wf_h + (size_t)(w * 32 + lane15) * DD + 16 * hi;
  const _Float16* b1p  = Wf_h + (size_t)(w * 32 + 16 + lane15) * DD + 16 * hi;

  v8f acc0 = {}, acc1 = {};
#pragma unroll 4
  for (int k0 = 0; k0 < DD; k0 += 32) {
    v16h a  = gfragA(arow + k0, hi);
    v16h b0 = *(const v16h*)(b0p + k0);
    v16h b1 = *(const v16h*)(b1p + k0);
    acc0 = wmma_f16(a, b0, acc0);
    acc1 = wmma_f16(a, b1, acc1);
  }
  {  // D-layout scatter to LDS with bias
    int Mb = 8 * hi;
    int c0 = w * 32 + lane15;
#pragma unroll
    for (int r = 0; r < 8; ++r) {
      outT[(Mb + r) * 132 + c0]      = acc0[r] + bf[c0];
      outT[(Mb + r) * 132 + c0 + 16] = acc1[r] + bf[c0 + 16];
    }
  }
  __syncthreads();
  {  // row mask + partial sum of squares
    int r = tid >> 3, cb = (tid & 7) * 16;
    int msk = mask[row0 + r];
    float s = 0.f;
#pragma unroll
    for (int j = 0; j < 16; ++j) {
      float v = msk ? 0.f : outT[r * 132 + cb + j];
      outT[r * 132 + cb + j] = v;
      s += v * v;
    }
    part[r][tid & 7] = s;
  }
  __syncthreads();
  if (tid < 16) {
    float s = 0.f;
#pragma unroll
    for (int j = 0; j < 8; ++j) s += part[tid][j];
    invn[tid] = 1.0f / fmaxf(sqrtf(s), 1e-12f);
  }
  __syncthreads();
  {
    int r = tid >> 3, cb = (tid & 7) * 16;
    float inv = invn[r];
#pragma unroll
    for (int j = 0; j < 16; ++j)
      Xu[(size_t)(row0 + r) * DQv + cb + j] = (_Float16)(outT[r * 132 + cb + j] * inv);
  }
}

// ---------------------------------------------------------------- sup = xn*Wg^T + bg, row mask -> supT (TRANSPOSED [B][512][2048] f16)
// No LDS at all: operands direct from global, transposed store is one v8h per tile
// (D-layout gives each lane 8 consecutive rows at a fixed output column).
__global__ void __launch_bounds__(128) k_sup(const _Float16* __restrict__ xn,
                                             const _Float16* __restrict__ Wg_h,
                                             const float* __restrict__ bg,
                                             const int* __restrict__ mask,
                                             _Float16* __restrict__ supT) {
  const int tid  = threadIdx.x;
  const int lane = tid & 31;
  const int w    = tid >> 5;
  const int row0 = blockIdx.x * 16;
  const int cw0  = blockIdx.y * 128 + w * 32;
  const int lane15 = lane & 15;
  const int hi     = lane >> 4;

  const _Float16* arow = xn + (size_t)(row0 + lane15) * DD;
  const _Float16* b0p  = Wg_h + (size_t)(cw0 + lane15) * DD + 16 * hi;
  const _Float16* b1p  = Wg_h + (size_t)(cw0 + 16 + lane15) * DD + 16 * hi;

  v8f acc0 = {}, acc1 = {};
#pragma unroll 4
  for (int k0 = 0; k0 < DD; k0 += 32) {
    v16h a  = gfragA(arow + k0, hi);
    v16h b0 = *(const v16h*)(b0p + k0);
    v16h b1 = *(const v16h*)(b1p + k0);
    acc0 = wmma_f16(a, b0, acc0);
    acc1 = wmma_f16(a, b1, acc1);
  }
  const int b  = row0 >> 11;          // batch
  const int nb = row0 & 2047;         // row within batch
  int msk[8];
#pragma unroll
  for (int r = 0; r < 8; ++r) msk[r] = mask[row0 + 8 * hi + r];

#pragma unroll
  for (int s = 0; s < 2; ++s) {
    const int col = cw0 + s * 16 + lane15;
    const float bgl = bg[col];
    const v8f& acc = s ? acc1 : acc0;
    v8h hv;
#pragma unroll
    for (int r = 0; r < 8; ++r)
      hv[r] = (_Float16)(msk[r] ? 0.f : (acc[r] + bgl));
    *(v8h*)(supT + ((size_t)(b * DD + col)) * NN + nb + 8 * hi) = hv;
  }
}

// ---------------------------------------------------------------- fused: S = Q*K^T, exp (no-max softmax, cos-sim bounded),
// out = P*supT / denom, residual. Barrier-free inner loop; only per-wave LDS P transpose.
__global__ void __launch_bounds__(128) k_attn(const float* __restrict__ x,
                                              const _Float16* __restrict__ Xu,
                                              const _Float16* __restrict__ supT,
                                              const int* __restrict__ mask,
                                              float* __restrict__ out) {
  __shared__ __align__(16) _Float16 Pt[4][16 * 34];  // per-wave P tile (padded)
  const int tid    = threadIdx.x;
  const int lane   = tid & 31;
  const int w      = tid >> 5;
  const int b      = blockIdx.x >> 7;
  const int m0     = (blockIdx.x & 127) << 4;
  const int bBase  = b * NN;
  const int lane15 = lane & 15;
  const int hi     = lane >> 4;

  // Q A-fragments from global (layout-exact contiguous loads)
  v16h qa[4];
  {
    const _Float16* qrow = Xu + (size_t)(bBase + m0 + lane15) * DQv;
#pragma unroll
    for (int ks = 0; ks < 4; ++ks) qa[ks] = gfragA(qrow + ks * 32, hi);
  }
  int rm[8];
#pragma unroll
  for (int r = 0; r < 8; ++r) rm[r] = mask[bBase + m0 + 8 * hi + r];

  v8f o[8] = {};
  float psum[8] = {0.f, 0.f, 0.f, 0.f, 0.f, 0.f, 0.f, 0.f};

  for (int n0 = 0; n0 < NN; n0 += 32) {
    // scores S[16x32]: K B-frags are 32B contiguous global loads
    v8f s0 = {}, s1 = {};
#pragma unroll
    for (int ks = 0; ks < 4; ++ks) {
      const _Float16* kb0 = Xu + (size_t)(bBase + n0 + lane15) * DQv + ks * 32 + hi * 16;
      const _Float16* kb1 = Xu + (size_t)(bBase + n0 + 16 + lane15) * DQv + ks * 32 + hi * 16;
      s0 = wmma_f16(qa[ks], *(const v16h*)kb0, s0);
      s1 = wmma_f16(qa[ks], *(const v16h*)kb1, s1);
    }
    const int cm0 = mask[bBase + n0 + lane15];
    const int cm1 = mask[bBase + n0 + 16 + lane15];
#pragma unroll
    for (int r = 0; r < 8; ++r) {
      float p0 = cm0 ? 0.f : __expf(s0[r]);
      float p1 = cm1 ? 0.f : __expf(s1[r]);
      psum[r] += p0 + p1;
      int M = 8 * hi + r;
      Pt[w][M * 34 + lane15]      = (_Float16)p0;
      Pt[w][M * 34 + 16 + lane15] = (_Float16)p1;
    }
    // per-wave private region; DS is in-order within a wave — wait + compiler fence
    asm volatile("s_wait_dscnt 0x0" ::: "memory");
    v16h pa = ldsA16x32(&Pt[w][0], 34, lane);
#pragma unroll
    for (int c = 0; c < 8; ++c) {
      const int colg = w * 128 + c * 16 + lane15;
      const _Float16* bp = supT + ((size_t)(b * DD + colg)) * NN + n0 + 16 * hi;
      o[c] = wmma_f16(pa, *(const v16h*)bp, o[c]);
    }
  }
  // row-sum reduction across the 16 lanes of each half (rows match C-layout ownership)
#pragma unroll
  for (int r = 0; r < 8; ++r) {
#pragma unroll
    for (int m = 1; m < 16; m <<= 1) psum[r] += __shfl_xor(psum[r], m, 32);
  }
#pragma unroll
  for (int c = 0; c < 8; ++c) {
#pragma unroll
    for (int r = 0; r < 8; ++r) {
      int M = 8 * hi + r;
      size_t idx = (size_t)(bBase + m0 + M) * DD + w * 128 + c * 16 + lane15;
      float add = rm[r] ? 0.f : (o[c][r] / psum[r]);
      out[idx] = x[idx] + add;
    }
  }
}

// ----------------------------------------------------------------
extern "C" void kernel_launch(void* const* d_in, const int* in_sizes, int n_in,
                              void* d_out, int out_size, void* d_ws, size_t ws_size,
                              hipStream_t stream) {
  const float* x     = (const float*)d_in[0];
  const float* gamma = (const float*)d_in[1];
  const float* beta  = (const float*)d_in[2];
  const float* Wf    = (const float*)d_in[3];
  const float* bf    = (const float*)d_in[4];
  const float* Wg    = (const float*)d_in[5];
  const float* bg    = (const float*)d_in[6];
  const int*   mask  = (const int*)d_in[7];
  float* out = (float*)d_out;

  // workspace layout (f16 buffers), ~38.4 MB total
  _Float16* xn_h = (_Float16*)d_ws;
  _Float16* Xu   = xn_h + (size_t)ROWS * DD;   // +16.8 MB
  _Float16* supT = Xu + (size_t)ROWS * DQv;    // +4.2 MB  (transposed [B][512][2048])
  _Float16* Wf_h = supT + (size_t)ROWS * DD;   // +16.8 MB
  _Float16* Wg_h = Wf_h + (size_t)DQv * DD;    // +128 KB

  k_prep<<<512, 256, 0, stream>>>(Wf, Wg, Wf_h, Wg_h);
  k_layernorm<<<ROWS, 128, 0, stream>>>(x, gamma, beta, xn_h);
  k_proj_norm<<<ROWS / 16, 128, 0, stream>>>(xn_h, Wf_h, bf, mask, Xu);
  k_sup<<<dim3(ROWS / 16, DD / 128), 128, 0, stream>>>(xn_h, Wg_h, bg, mask, supT);
  k_attn<<<ROWS / 16, 128, 0, stream>>>(x, Xu, supT, mask, out);
}
